// RNN_63127429317309
// MI455X (gfx1250) — matrix operand order repaired and checked by reading, not compiled, not run on hardware
//
#include <hip/hip_runtime.h>

// ---------------------------------------------------------------------------
// Fused 3-layer LSTM (H=64, T=2048, B=256) + final projection, for gfx1250.
// One workgroup (128 thr = 4 waves) per 16-row batch tile; wave q owns gate
// columns [16q,16q+16) of each of the 4 gates. All weight B-tiles live in
// VGPRs across the whole time loop; h state is exchanged via 12KB of LDS.
// ---------------------------------------------------------------------------

typedef _Float16 v16h __attribute__((ext_vector_type(16)));
typedef _Float16 v8h  __attribute__((ext_vector_type(8)));
typedef float    v8f  __attribute__((ext_vector_type(8)));

#define T_STEPS 2048
#define NBLK    16   // 256 batch rows / 16 per workgroup

#define WMMA(a, b, c) \
    __builtin_amdgcn_wmma_f32_16x16x32_f16(false, (a), false, (b), (short)0, (c), false, false)

__device__ __forceinline__ float fast_sigmoid(float x) {
    float e = __builtin_amdgcn_exp2f(-1.4426950408889634f * x);
    return __builtin_amdgcn_rcpf(1.0f + e);
}
__device__ __forceinline__ float fast_tanh(float x) {
    float e = __builtin_amdgcn_exp2f(-2.8853900817779268f * x);
    return 2.0f * __builtin_amdgcn_rcpf(1.0f + e) - 1.0f;
}

__device__ __forceinline__ v8f splat8(float x) {
    v8f r;
#pragma unroll
    for (int i = 0; i < 8; ++i) r[i] = x;
    return r;
}

__device__ __forceinline__ v16h zero16() {
    v16h r;
#pragma unroll
    for (int i = 0; i < 16; ++i) r[i] = (_Float16)0.0f;
    return r;
}

// Load a 32(K) x 16(N) f16 B-operand tile from a row-major f32 weight matrix
// W[rows][stride] covering rows [row_base,row_base+16), cols [k_base,k_base+32).
// ISA B layout: lane n (0..15) holds column row_base+n, K=k_base+{0..7,16..23};
// lane n+16 holds K=k_base+{8..15,24..31} (2 halves per VGPR).
__device__ __forceinline__ v16h load_btile(const float* W, int stride,
                                           int row_base, int k_base, int lane) {
    const int n  = lane & 15;
    const int kh = (lane >> 4) << 3;   // 0 or 8
    const float* src = W + (row_base + n) * stride + k_base + kh;
    v16h b;
#pragma unroll
    for (int i = 0; i < 16; ++i) b[i] = (_Float16)src[i < 8 ? i : i + 8];
    return b;
}

// Load a 16(M) x 32(K) f16 A-operand tile from an LDS row-major [16][64] h tile.
// ISA A layout: lane m holds row m, K = kb*32 + {0..7, 16..23};
// lane m+16 holds row m, K = kb*32 + {8..15, 24..31}.
__device__ __forceinline__ v16h load_atile(const _Float16* h, int kb, int lane) {
    const int m  = lane & 15;
    const int hi = (lane >> 4) << 3;   // 0 or 8
    const _Float16* p = h + m * 64 + kb * 32 + hi;
    v8h lo = *(const v8h*)p;
    v8h hh = *(const v8h*)(p + 16);
    return __builtin_shufflevector(lo, hh,
        0, 1, 2, 3, 4, 5, 6, 7, 8, 9, 10, 11, 12, 13, 14, 15);
}

// Gate activations + cell/hidden update + f16 h store to LDS.
// C-tile layout: lane n holds (M=v, N=n) for v=0..7; lane n+16 holds M=v+8.
__device__ __forceinline__ void lstm_tail(const v8f g[4], v8f& cst,
                                          _Float16* hdst, int q, int lane) {
    const int n  = lane & 15;
    const int rb = (lane >> 4) << 3;   // 0 or 8
#pragma unroll
    for (int v = 0; v < 8; ++v) {
        float iv = fast_sigmoid(g[0][v]);
        float fv = fast_sigmoid(g[1][v]);
        float gv = fast_tanh(g[2][v]);
        float ov = fast_sigmoid(g[3][v]);
        float cv = fv * cst[v] + iv * gv;
        cst[v] = cv;
        float hv = ov * fast_tanh(cv);
        hdst[(rb + v) * 64 + q * 16 + n] = (_Float16)hv;
    }
}

__global__ __launch_bounds__(128, 1)
void lstm3_fused(const float* __restrict__ x,
                 const float* __restrict__ Wih0, const float* __restrict__ Whh0,
                 const float* __restrict__ bih0, const float* __restrict__ bhh0,
                 const float* __restrict__ Wih1, const float* __restrict__ Whh1,
                 const float* __restrict__ bih1, const float* __restrict__ bhh1,
                 const float* __restrict__ Wih2, const float* __restrict__ Whh2,
                 const float* __restrict__ bih2, const float* __restrict__ bhh2,
                 const float* __restrict__ Wout, const float* __restrict__ bout,
                 float* __restrict__ out)
{
    // double-buffered per-layer h tiles: [layer][parity][row 16][col 64]
    __shared__ _Float16 hbuf[3][2][16][64];

    const int tid  = threadIdx.x;
    const int lane = tid & 31;
    const int q    = tid >> 5;        // wave id == gate-column quarter
    const int n    = lane & 15;
    const int b0   = blockIdx.x * 16; // batch tile base

    // zero initial h state (both parities, all layers)
    for (int i = tid; i < 3 * 2 * 16 * 64; i += 128)
        (&hbuf[0][0][0][0])[i] = (_Float16)0.0f;

    const float* WhhL[3] = {Whh0, Whh1, Whh2};
    const float* WihL[2] = {Wih1, Wih2};
    const float* bihL[3] = {bih0, bih1, bih2};
    const float* bhhL[3] = {bhh0, bhh1, bhh2};

    // ---- hoist every B-operand weight tile into VGPRs (f32 -> f16) ----
    v16h Bhh[3][4][2];   // [layer][gate][k-block]
    v16h Bih[2][4][2];   // layers 1,2
    v16h Bih0[4];        // layer 0, K padded 2 -> 32
#pragma unroll
    for (int l = 0; l < 3; ++l)
#pragma unroll
        for (int G = 0; G < 4; ++G)
#pragma unroll
            for (int kb = 0; kb < 2; ++kb)
                Bhh[l][G][kb] = load_btile(WhhL[l], 64, G * 64 + q * 16, kb * 32, lane);
#pragma unroll
    for (int l = 0; l < 2; ++l)
#pragma unroll
        for (int G = 0; G < 4; ++G)
#pragma unroll
            for (int kb = 0; kb < 2; ++kb)
                Bih[l][G][kb] = load_btile(WihL[l], 64, G * 64 + q * 16, kb * 32, lane);
#pragma unroll
    for (int G = 0; G < 4; ++G) {
        v16h b = zero16();
        if (lane < 16) {             // only K=0,1 are real; lanes 16-31 all zero
            const float* src = Wih0 + (G * 64 + q * 16 + n) * 2;
            b[0] = (_Float16)src[0];
            b[1] = (_Float16)src[1];
        }
        Bih0[G] = b;
    }

    // per-lane bias for this wave's gate columns (b_ih + b_hh)
    float bias[3][4];
#pragma unroll
    for (int l = 0; l < 3; ++l)
#pragma unroll
        for (int G = 0; G < 4; ++G) {
            const int col = G * 64 + q * 16 + n;
            bias[l][G] = bihL[l][col] + bhhL[l][col];
        }

    v8f cst0 = splat8(0.0f), cst1 = splat8(0.0f), cst2 = splat8(0.0f);

    __syncthreads();

    for (int t = 0; t < T_STEPS; ++t) {
        const int cur = t & 1, prev = cur ^ 1;

        // ---------------- layer 0 (x_t is [16 x 2], K-padded to 32) --------
        {
            v16h ax = zero16();
            if (lane < 16) {
                const float* xs = x + (size_t)(b0 + n) * (T_STEPS * 2) + (size_t)t * 2;
                ax[0] = (_Float16)xs[0];
                ax[1] = (_Float16)xs[1];
            }
            v16h ah0 = load_atile(&hbuf[0][prev][0][0], 0, lane);
            v16h ah1 = load_atile(&hbuf[0][prev][0][0], 1, lane);
            v8f g[4];
#pragma unroll
            for (int G = 0; G < 4; ++G) {
                v8f acc = splat8(bias[0][G]);
                acc = WMMA(ax,  Bih0[G],      acc);
                acc = WMMA(ah0, Bhh[0][G][0], acc);
                acc = WMMA(ah1, Bhh[0][G][1], acc);
                g[G] = acc;
            }
            lstm_tail(g, cst0, &hbuf[0][cur][0][0], q, lane);
        }
        __syncthreads();

        // ---------------- layers 1 and 2 -----------------------------------
#pragma unroll
        for (int l = 1; l <= 2; ++l) {
            v16h ai0 = load_atile(&hbuf[l - 1][cur][0][0], 0, lane);
            v16h ai1 = load_atile(&hbuf[l - 1][cur][0][0], 1, lane);
            v16h ah0 = load_atile(&hbuf[l][prev][0][0],    0, lane);
            v16h ah1 = load_atile(&hbuf[l][prev][0][0],    1, lane);
            v8f g[4];
#pragma unroll
            for (int G = 0; G < 4; ++G) {
                v8f acc = splat8(bias[l][G]);
                acc = WMMA(ai0, Bih[l - 1][G][0], acc);
                acc = WMMA(ai1, Bih[l - 1][G][1], acc);
                acc = WMMA(ah0, Bhh[l][G][0],     acc);
                acc = WMMA(ah1, Bhh[l][G][1],     acc);
                g[G] = acc;
            }
            v8f& cst = (l == 1) ? cst1 : cst2;
            lstm_tail(g, cst, &hbuf[l][cur][0][0], q, lane);
            __syncthreads();
        }
    }

    // ---------------- final projection: out = h2_last @ Wout^T + bout ------
    const int lastp = (T_STEPS - 1) & 1;
    if (tid < 16 * 11) {
        const int r = tid / 11, k = tid % 11;
        float acc = bout[k];
#pragma unroll 8
        for (int h = 0; h < 64; ++h)
            acc += (float)hbuf[2][lastp][r][h] * Wout[k * 64 + h];
        out[(b0 + r) * 11 + k] = acc;
    }
}

extern "C" void kernel_launch(void* const* d_in, const int* in_sizes, int n_in,
                              void* d_out, int out_size, void* d_ws, size_t ws_size,
                              hipStream_t stream) {
    const float* x    = (const float*)d_in[0];
    const float* Wih0 = (const float*)d_in[1];
    const float* Whh0 = (const float*)d_in[2];
    const float* bih0 = (const float*)d_in[3];
    const float* bhh0 = (const float*)d_in[4];
    const float* Wih1 = (const float*)d_in[5];
    const float* Whh1 = (const float*)d_in[6];
    const float* bih1 = (const float*)d_in[7];
    const float* bhh1 = (const float*)d_in[8];
    const float* Wih2 = (const float*)d_in[9];
    const float* Whh2 = (const float*)d_in[10];
    const float* bih2 = (const float*)d_in[11];
    const float* bhh2 = (const float*)d_in[12];
    const float* Wout = (const float*)d_in[13];
    const float* bout = (const float*)d_in[14];

    lstm3_fused<<<NBLK, 128, 0, stream>>>(x,
        Wih0, Whh0, bih0, bhh0,
        Wih1, Whh1, bih1, bhh1,
        Wih2, Whh2, bih2, bhh2,
        Wout, bout, (float*)d_out);
}
